// rhs_base_39470749450352
// MI455X (gfx1250) — matrix operand order repaired and checked by reading, not compiled, not run on hardware
//
#include <hip/hip_runtime.h>
#include <hip/hip_fp16.h>

typedef __attribute__((ext_vector_type(16))) _Float16 v16h;
typedef __attribute__((ext_vector_type(4)))  _Float16 v4h;
typedef __attribute__((ext_vector_type(8)))  float    v8f;
typedef __attribute__((ext_vector_type(4)))  float    v4f;

#define DIM   256
#define NEXP  8
#define MT    32      // rows per block tile
#define KC    32      // K chunk per LDS stage
#define YSTR  264     // padded f16 row stride for Y/H tiles (even, 8B-alignable)
#define WSTR  40      // padded f16 row stride for W chunk  (even, 8B-alignable)

__device__ __forceinline__ float sigmoidf_(float x) {
    return 1.0f / (1.0f + __expf(-x));
}

__device__ __forceinline__ float fast_tanh(float x) {
#if __has_builtin(__builtin_amdgcn_tanhf)
    return __builtin_amdgcn_tanhf(x);          // v_tanh_f32 (gfx1250 trans op)
#else
    float e = __expf(2.0f * x);                // branch-free fallback
    return 1.0f - 2.0f / (e + 1.0f);
#endif
}

// ---------------- fragment loaders (wave32 WMMA; this code's contiguous k order, same for A and B) ----------------
// A (16x32 f16): lanes 0-15 -> M=lane, K = 0..15 (element s -> K=s)
//                lanes 16-31 -> M=lane-16, K = 16..31.
__device__ __forceinline__ v16h load_A(const _Float16* base, int m0, int k0,
                                       int laneM, int hi) {
    const _Float16* rp = base + (m0 + laneM) * YSTR + k0 + (hi ? 16 : 0);
    v16h a;
#pragma unroll
    for (int v = 0; v < 4; ++v) {
        a[2 * v]         = rp[2 * v];
        a[2 * v + 1]     = rp[2 * v + 1];
        a[8 + 2 * v]     = rp[8 + 2 * v];
        a[8 + 2 * v + 1] = rp[8 + 2 * v + 1];
    }
    return a;
}

// B (32x16 f16): lane = N (mod 16); lanes 0-15 hold K=0..15 (2/VGPR),
//                lanes 16-31 hold K=16..31.  B[k][n] = Wt[n][k] (Wt row-major [n][k]).
__device__ __forceinline__ v16h load_B(const _Float16* wt, int n0,
                                       int laneM, int hi) {
    const _Float16* cp = wt + (n0 + laneM) * WSTR + (hi ? 16 : 0);
    v16h b;
#pragma unroll
    for (int v = 0; v < 8; ++v) {
        b[2 * v]     = cp[2 * v];
        b[2 * v + 1] = cp[2 * v + 1];
    }
    return b;
}

// stage a 256x32 f32 weight chunk into f16 LDS, float4 loads / 8B LDS stores
__device__ __forceinline__ void stage_W(const float* __restrict__ We, int k0,
                                        _Float16* wt, int tid) {
#pragma unroll
    for (int j = 0; j < 8; ++j) {              // 2048 float4 / 256 threads
        int g  = j * 256 + tid;
        int nn = g >> 3;                       // row (N)
        int kk = (g & 7) * 4;                  // col (K) within chunk
        v4f w = *(const v4f*)(We + nn * DIM + k0 + kk);
        v4h h = { (_Float16)w.x, (_Float16)w.y, (_Float16)w.z, (_Float16)w.w };
        *(v4h*)(wt + nn * WSTR + kk) = h;
    }
}

// ---------------- kernels ----------------
__global__ void zero_counts_k(int* counts) {
    if (threadIdx.x < NEXP) counts[threadIdx.x] = 0;
}

__global__ void assign_k(const float* __restrict__ t, int* __restrict__ counts,
                         int* __restrict__ bucket, int n) {
    int i = blockIdx.x * blockDim.x + threadIdx.x;
    if (i >= n) return;
    int e = (int)(t[i] * 8.0f);                // t / (1/8), truncation toward zero
    e = e > (NEXP - 1) ? (NEXP - 1) : (e < 0 ? 0 : e);
    int pos = atomicAdd(&counts[e], 1);
    bucket[e * n + pos] = i;
}

__global__ __launch_bounds__(256, 1) void mlp_k(
    const float* __restrict__ y,  const float* __restrict__ W1,
    const float* __restrict__ b1, const float* __restrict__ W2,
    const float* __restrict__ b2, const float* __restrict__ scales,
    const float* __restrict__ shifta, const float* __restrict__ shiftb,
    const int* __restrict__ counts, const int* __restrict__ bucket,
    float* __restrict__ out, int n) {

    const int e   = blockIdx.y;
    const int cnt = counts[e];
    const int m0g = blockIdx.x * MT;
    if (m0g >= cnt) return;

    __shared__ _Float16 Yt[MT * YSTR];
    __shared__ _Float16 Ht[MT * YSTR];
    __shared__ _Float16 Wt[DIM * WSTR];
    __shared__ int rIdx[MT];

    const int tid   = threadIdx.x;
    const int wave  = tid >> 5;
    const int lane  = tid & 31;
    const int laneM = lane & 15;
    const int hi    = lane >> 4;
    const int nBase = wave * 32;               // each wave owns 32 N-columns

    if (tid < MT) {
        int gm = m0g + tid;
        rIdx[tid] = (gm < cnt) ? bucket[e * n + gm] : -1;
    }
    __syncthreads();

    // stage Y tile (f32 -> f16): float4 loads, 8B LDS stores.
    // 8192 elems = 2048 float4; iter j covers 4 rows (64 float4/row).
#pragma unroll
    for (int j = 0; j < 8; ++j) {
        int g   = j * 256 + tid;
        int r   = g >> 6;                      // tile row
        int c   = (g & 63) * 4;                // column
        int row = rIdx[r];
        v4f yv = {0.0f, 0.0f, 0.0f, 0.0f};
        if (row >= 0) yv = *(const v4f*)(y + (long)row * DIM + c);
        v4h h = { (_Float16)yv.x, (_Float16)yv.y, (_Float16)yv.z, (_Float16)yv.w };
        *(v4h*)(Yt + r * YSTR + c) = h;
    }

    const float* W1e = W1 + (long)e * DIM * DIM;
    const float* W2e = W2 + (long)e * DIM * DIM;

    // ---------------- layer 1: H = tanh(Y * W1^T + b1) ----------------
    v8f acc[2][2];
#pragma unroll
    for (int mt = 0; mt < 2; ++mt)
#pragma unroll
        for (int nt = 0; nt < 2; ++nt) acc[mt][nt] = (v8f){0,0,0,0,0,0,0,0};

    for (int k0 = 0; k0 < DIM; k0 += KC) {
        __syncthreads();                       // protect Wt reuse
        stage_W(W1e, k0, Wt, tid);
        __syncthreads();

        v16h a0  = load_A(Yt, 0,  k0, laneM, hi);
        v16h a1  = load_A(Yt, 16, k0, laneM, hi);
        v16h bb0 = load_B(Wt, nBase,      laneM, hi);
        v16h bb1 = load_B(Wt, nBase + 16, laneM, hi);
        acc[0][0] = __builtin_amdgcn_wmma_f32_16x16x32_f16(false, a0, false, bb0, (short)0, acc[0][0], false, false);
        acc[0][1] = __builtin_amdgcn_wmma_f32_16x16x32_f16(false, a0, false, bb1, (short)0, acc[0][1], false, false);
        acc[1][0] = __builtin_amdgcn_wmma_f32_16x16x32_f16(false, a1, false, bb0, (short)0, acc[1][0], false, false);
        acc[1][1] = __builtin_amdgcn_wmma_f32_16x16x32_f16(false, a1, false, bb1, (short)0, acc[1][1], false, false);
    }

    // bias + tanh, write H tile (C/D layout: lane=N, VGPR r -> M=r (+8 for hi half))
    __syncthreads();
#pragma unroll
    for (int mt = 0; mt < 2; ++mt) {
#pragma unroll
        for (int nt = 0; nt < 2; ++nt) {
            int nn = nBase + nt * 16 + laneM;
            float bias = b1[e * DIM + nn];
#pragma unroll
            for (int r = 0; r < 8; ++r) {
                int mm = mt * 16 + hi * 8 + r;
                float h = fast_tanh(acc[mt][nt][r] + bias);
                Ht[mm * YSTR + nn] = (_Float16)h;
            }
        }
    }

    // ---------------- layer 2: F = H * W2^T + b2 ----------------
#pragma unroll
    for (int mt = 0; mt < 2; ++mt)
#pragma unroll
        for (int nt = 0; nt < 2; ++nt) acc[mt][nt] = (v8f){0,0,0,0,0,0,0,0};

    for (int k0 = 0; k0 < DIM; k0 += KC) {
        __syncthreads();
        stage_W(W2e, k0, Wt, tid);
        __syncthreads();

        v16h a0  = load_A(Ht, 0,  k0, laneM, hi);
        v16h a1  = load_A(Ht, 16, k0, laneM, hi);
        v16h bb0 = load_B(Wt, nBase,      laneM, hi);
        v16h bb1 = load_B(Wt, nBase + 16, laneM, hi);
        acc[0][0] = __builtin_amdgcn_wmma_f32_16x16x32_f16(false, a0, false, bb0, (short)0, acc[0][0], false, false);
        acc[0][1] = __builtin_amdgcn_wmma_f32_16x16x32_f16(false, a0, false, bb1, (short)0, acc[0][1], false, false);
        acc[1][0] = __builtin_amdgcn_wmma_f32_16x16x32_f16(false, a1, false, bb0, (short)0, acc[1][0], false, false);
        acc[1][1] = __builtin_amdgcn_wmma_f32_16x16x32_f16(false, a1, false, bb1, (short)0, acc[1][1], false, false);
    }

    // ---------------- epilogue ----------------
    float a_s = -sigmoidf_(shifta[0]);         // EIGINIT + sig*(EIGMIN-EIGINIT)
    float b_s =  sigmoidf_(shiftb[0]);         // EIGINIT + sig*(EIGMAX-EIGINIT)
    float c1  = 0.5f * (b_s - a_s);
    float c2  = 0.5f * (a_s + b_s);

#pragma unroll
    for (int mt = 0; mt < 2; ++mt) {
#pragma unroll
        for (int nt = 0; nt < 2; ++nt) {
            int nn = nBase + nt * 16 + laneM;
            float sc    = sigmoidf_(scales[nn]);
            float bias2 = b2[e * DIM + nn];
#pragma unroll
            for (int r = 0; r < 8; ++r) {
                int mm  = mt * 16 + hi * 8 + r;
                int row = rIdx[mm];
                if (row < 0) continue;
                float f  = sc * (acc[mt][nt][r] + bias2);
                float yv = y[(long)row * DIM + nn];   // full-precision y
                out[(long)row * DIM + nn] = c1 * f + c2 * yv;
            }
        }
    }
}

extern "C" void kernel_launch(void* const* d_in, const int* in_sizes, int n_in,
                              void* d_out, int out_size, void* d_ws, size_t ws_size,
                              hipStream_t stream) {
    const float* t      = (const float*)d_in[0];
    const float* y      = (const float*)d_in[1];
    const float* W1     = (const float*)d_in[2];
    const float* b1     = (const float*)d_in[3];
    const float* W2     = (const float*)d_in[4];
    const float* b2     = (const float*)d_in[5];
    const float* scales = (const float*)d_in[6];
    const float* shifta = (const float*)d_in[7];
    const float* shiftb = (const float*)d_in[8];
    float* out = (float*)d_out;

    const int n = in_sizes[0];                 // batch size (32768)

    int* counts = (int*)d_ws;                  // 8 ints
    int* bucket = (int*)((char*)d_ws + 64);    // NEXP * n ints

    zero_counts_k<<<1, 32, 0, stream>>>(counts);
    assign_k<<<(n + 255) / 256, 256, 0, stream>>>(t, counts, bucket, n);

    dim3 grid((n + MT - 1) / MT, NEXP);
    mlp_k<<<grid, 256, 0, stream>>>(y, W1, b1, W2, b2, scales, shifta, shiftb,
                                    counts, bucket, out, n);
}
